// Head_55671366091048
// MI455X (gfx1250) — matrix operand order repaired and checked by reading, not compiled, not run on hardware
//
#include <hip/hip_runtime.h>
#include <hip/hip_bf16.h>
#include <cstdint>

// ---------------------------------------------------------------------------
// MI455X (gfx1250) attention head:
//   q = x Wq^T ; k = y Wk^T ; vT = (y Wv^T)^T     (bf16 WMMA, f32 accum)
//   out = mask(q k^T * d^-1/2) v                  (flash-style streaming)
// CDNA5 paths: v_wmma_f32_16x16x32_bf16, tensor_load_to_lds (TDM, in-order
// completion + s_wait_tensorcnt thresholds for K double-buffering),
// global_load_async_to_lds_b128 (ASYNCcnt), global_prefetch, 264KB WGP LDS.
// ---------------------------------------------------------------------------

typedef __attribute__((ext_vector_type(16))) __bf16   v16bf;
typedef __attribute__((ext_vector_type(2)))  __bf16   v2bf;
typedef __attribute__((ext_vector_type(8)))  float    v8f;
typedef __attribute__((ext_vector_type(4)))  uint32_t v4u;
typedef __attribute__((ext_vector_type(8)))  uint32_t v8u;

#define NXR   8192
#define NYR   8192
#define CIN   2048
#define DHEAD 512

// f32x2 -> bf16x2; native packed convert when declared, else fptrunc (RNE).
static __device__ __forceinline__ v2bf pack_bf16(float a, float b) {
#if __has_builtin(__builtin_amdgcn_cvt_pk_bf16_f32)
  return __builtin_amdgcn_cvt_pk_bf16_f32(a, b);
#else
  v2bf r; r[0] = (__bf16)a; r[1] = (__bf16)b; return r;
#endif
}

// Async global->LDS copy, 16B per lane, tracked by ASYNCcnt (CDNA5).
static __device__ __forceinline__ void async_load_b128(uint32_t lds_off,
                                                       const void* gptr) {
  asm volatile("global_load_async_to_lds_b128 %0, %1, off"
               :: "v"(lds_off), "v"(gptr) : "memory");
}

static __device__ __forceinline__ void wait_async_all() {
#if __has_builtin(__builtin_amdgcn_s_wait_asynccnt)
  __builtin_amdgcn_s_wait_asynccnt(0);
#else
  asm volatile("s_wait_asynccnt 0" ::: "memory");
#endif
}

static __device__ __forceinline__ void wait_tensor0() {
#if __has_builtin(__builtin_amdgcn_s_wait_tensorcnt)
  __builtin_amdgcn_s_wait_tensorcnt(0);
#else
  asm volatile("s_wait_tensorcnt 0" ::: "memory");
#endif
}

// TDM ops from one wave complete in order: waiting <=1 with [V, Knext]
// outstanding proves V (and everything issued before it) has landed.
static __device__ __forceinline__ void wait_tensor1() {
#if __has_builtin(__builtin_amdgcn_s_wait_tensorcnt)
  __builtin_amdgcn_s_wait_tensorcnt(1);
#else
  asm volatile("s_wait_tensorcnt 1" ::: "memory");
#endif
}

// Tensor Data Mover: 2-D bf16 tile (tile_d0 x tile_d1 elements) global -> LDS.
// D# bitfields per CDNA5 ISA ch.8; groups 2/3 zero (dims 2..4 unused).
static __device__ __forceinline__ void tdm_load_2d_bf16(
    uint32_t lds_off, const void* gptr,
    uint32_t tensor_d0, uint32_t tensor_d1, uint32_t stride_d0,
    uint32_t tile_d0, uint32_t tile_d1) {
  const uint64_t ga = (uint64_t)(uintptr_t)gptr;
  v4u g0;
  g0[0] = 1u;                                                // count=1 (valid)
  g0[1] = lds_off;                                           // lds_addr (bytes)
  g0[2] = (uint32_t)ga;                                      // global_addr lo
  g0[3] = (uint32_t)((ga >> 32) & 0x01FFFFFFu) | (2u << 30); // ga[56:32], type=2
  v8u g1;
  g1[0] = 1u << 16;                                          // data_size=1 (2B)
  g1[1] = (tensor_d0 & 0xFFFFu) << 16;
  g1[2] = (tensor_d0 >> 16) | ((tensor_d1 & 0xFFFFu) << 16);
  g1[3] = (tensor_d1 >> 16) | ((tile_d0 & 0xFFFFu) << 16);
  g1[4] = tile_d1 & 0xFFFFu;                                 // tile_dim1
  g1[5] = stride_d0;                                         // dim0 stride lo32
  g1[6] = 0;
  g1[7] = 0;
  const v4u gz = {0u, 0u, 0u, 0u};
  asm volatile("tensor_load_to_lds %0, %1, %2, %3"
               :: "s"(g0), "s"(g1), "s"(gz), "s"(gz)
               : "memory");
}

// A-matrix fragment (16x32 bf16, MxK), ISA 7.12.2 layout.
static __device__ __forceinline__ v16bf load_frag_a(const __bf16* p, int ld,
                                                    int m0, int k0) {
  const int lane = threadIdx.x & 31;
  const int r    = lane & 15;
  const int half = lane >> 4;
  const __bf16* row = p + (size_t)(m0 + r) * ld + k0 + half * 8;
  v16bf a;
#pragma unroll
  for (int v = 0; v < 4; ++v) { a[2*v]   = row[2*v];      a[2*v+1]   = row[2*v+1]; }
#pragma unroll
  for (int v = 0; v < 4; ++v) { a[8+2*v] = row[16 + 2*v]; a[8+2*v+1] = row[16 + 2*v + 1]; }
  return a;
}

// B-matrix fragment (32x16 bf16, KxN), data stored [n][k] (k contiguous).
static __device__ __forceinline__ v16bf load_frag_b_nk(const __bf16* p, int ld,
                                                       int n0, int k0) {
  const int lane = threadIdx.x & 31;
  const int n    = lane & 15;
  const int half = lane >> 4;
  const __bf16* row = p + (size_t)(n0 + n) * ld + k0 + half * 8;
  v16bf b;
#pragma unroll
  for (int v = 0; v < 8; ++v) { b[2*v] = row[(v >> 2) * 16 + 2*(v & 3)]; b[2*v+1] = row[(v >> 2) * 16 + 2*(v & 3) + 1]; }
  return b;
}

// ---------------------------------------------------------------------------
// Kernel 1: C = A[M,CIN](f32) @ W[DHEAD,CIN](f32)^T -> bf16 (opt. transposed).
// 64x64 tile per 256-thread workgroup, BK=64 (4 WMMA per wave per stage).
// ---------------------------------------------------------------------------
template <int TRANS>
__global__ __launch_bounds__(256) void proj_gemm_bf16(
    const float* __restrict__ A, const float* __restrict__ W,
    __bf16* __restrict__ out) {
  constexpr int M = NXR;      // == NYR
  constexpr int K = CIN;
  constexpr int N = DHEAD;

  __shared__ __bf16 As[64 * 64];
  __shared__ __bf16 Ws[64 * 64];

  const int tid  = threadIdx.x;
  const int wave = tid >> 5;
  const int lane = tid & 31;
  const int m0   = blockIdx.x * 64;
  const int n0   = blockIdx.y * 64;
  const int mt   = wave >> 1;          // 0..3
  const int nt0  = (wave & 1) * 2;     // 0 or 2

  v8f acc0 = {}; v8f acc1 = {};

  const int sr = tid >> 2;             // staging row 0..63
  const int sk = (tid & 3) * 16;       // staging k-chunk 0,16,32,48

  const float* ga = A + (size_t)(m0 + sr) * K + sk;
  const float* gw = W + (size_t)(n0 + sr) * K + sk;

  for (int kb = 0; kb < K; kb += 64) {
    if (kb + 64 < K) {                 // prefetch next K-tile
      __builtin_prefetch(ga + kb + 64, 0, 1);
      __builtin_prefetch(gw + kb + 64, 0, 1);
    }
    {
      const float* g = ga + kb;
      float4 f0 = *(const float4*)(g);     float4 f1 = *(const float4*)(g + 4);
      float4 f2 = *(const float4*)(g + 8); float4 f3 = *(const float4*)(g + 12);
      alignas(16) v2bf t[8] = {pack_bf16(f0.x, f0.y), pack_bf16(f0.z, f0.w),
                               pack_bf16(f1.x, f1.y), pack_bf16(f1.z, f1.w),
                               pack_bf16(f2.x, f2.y), pack_bf16(f2.z, f2.w),
                               pack_bf16(f3.x, f3.y), pack_bf16(f3.z, f3.w)};
      *(uint4*)&As[sr * 64 + sk]     = ((const uint4*)t)[0];
      *(uint4*)&As[sr * 64 + sk + 8] = ((const uint4*)t)[1];
    }
    {
      const float* g = gw + kb;
      float4 f0 = *(const float4*)(g);     float4 f1 = *(const float4*)(g + 4);
      float4 f2 = *(const float4*)(g + 8); float4 f3 = *(const float4*)(g + 12);
      alignas(16) v2bf t[8] = {pack_bf16(f0.x, f0.y), pack_bf16(f0.z, f0.w),
                               pack_bf16(f1.x, f1.y), pack_bf16(f1.z, f1.w),
                               pack_bf16(f2.x, f2.y), pack_bf16(f2.z, f2.w),
                               pack_bf16(f3.x, f3.y), pack_bf16(f3.z, f3.w)};
      *(uint4*)&Ws[sr * 64 + sk]     = ((const uint4*)t)[0];
      *(uint4*)&Ws[sr * 64 + sk + 8] = ((const uint4*)t)[1];
    }
    __syncthreads();

#pragma unroll
    for (int ks = 0; ks < 64; ks += 32) {
      v16bf af  = load_frag_a   (As, 64, mt * 16, ks);
      v16bf bf0 = load_frag_b_nk(Ws, 64, nt0 * 16, ks);
      v16bf bf1 = load_frag_b_nk(Ws, 64, (nt0 + 1) * 16, ks);
      acc0 = __builtin_amdgcn_wmma_f32_16x16x32_bf16(false, af, false, bf0,
                                                     (short)0, acc0, false, false);
      acc1 = __builtin_amdgcn_wmma_f32_16x16x32_bf16(false, af, false, bf1,
                                                     (short)0, acc1, false, false);
    }
    __syncthreads();
  }

  // store C tiles: VGPR v, lane l -> m = v + (l>>4)*8, n = lane&15
  const int half = lane >> 4;
  const int nl   = lane & 15;
  const int c0   = n0 + nt0 * 16 + nl;
  const int c1   = c0 + 16;
#pragma unroll
  for (int v = 0; v < 8; ++v) {
    const int m = m0 + mt * 16 + v + half * 8;
    v2bf p = pack_bf16(acc0[v], acc1[v]);
    if (TRANS) {                        // out[N][M] = C^T
      out[(size_t)c0 * M + m] = p[0];
      out[(size_t)c1 * M + m] = p[1];
    } else {                            // out[M][N]
      out[(size_t)m * N + c0] = p[0];
      out[(size_t)m * N + c1] = p[1];
    }
  }
}

// ---------------------------------------------------------------------------
// Kernel 2: streaming causal attention. One 64-row Q block per workgroup
// (512 threads = 16 waves). K tiles double-buffered: TDM streams K(jb+1)
// behind compute; V(jb) correctness via in-order TDM + s_wait_tensorcnt 1.
// LDS: 64KB Q + 2x64KB K + 64KB V^T + 8KB S = 264KB of the 320KB WGP pool.
// ---------------------------------------------------------------------------
__global__ __launch_bounds__(512) void attn_stream(
    const __bf16* __restrict__ Q, const __bf16* __restrict__ K,
    const __bf16* __restrict__ Vt, float* __restrict__ out) {
  __shared__ __bf16 Qs [64 * DHEAD];     // [m][d]
  __shared__ __bf16 Ks [2][64 * DHEAD];  // [j][d], double-buffered
  __shared__ __bf16 Vts[DHEAD * 64];     // [d][j]
  __shared__ __bf16 Ss [64 * 64];        // [m][j]

  const int tid  = threadIdx.x;
  const int wave = tid >> 5;
  const int lane = tid & 31;
  const int ib   = blockIdx.x;
  const int row0 = ib * 64;

  const uint32_t qs_base = (uint32_t)(uintptr_t)Qs;
  const uint32_t ks_base0 = (uint32_t)(uintptr_t)&Ks[0][0];
  const uint32_t ks_base1 = (uint32_t)(uintptr_t)&Ks[1][0];
  const uint32_t vs_base = (uint32_t)(uintptr_t)Vts;

  // Preload K(0) via TDM while Q streams in via per-lane async copies.
  if (wave == 0)
    tdm_load_2d_bf16(ks_base0, K, DHEAD, NYR, DHEAD, DHEAD, 64);
  {
    const char* src = (const char*)(Q + (size_t)row0 * DHEAD);
#pragma unroll
    for (int i = 0; i < 8; ++i) {
      const uint32_t off = (uint32_t)(tid + i * 512) * 16u;
      async_load_b128(qs_base + off, src + off);
    }
  }

  // phase-2 ownership: wave -> 16 rows (mt2) x 128 cols (ng)
  const int mt2   = wave & 3;
  const int ng    = wave >> 2;
  const int ncol0 = ng * 128;

  v8f acc[8];
#pragma unroll
  for (int i = 0; i < 8; ++i) acc[i] = (v8f){};

  const float scale = 0.04419417382415922f;  // 512^-0.5

  for (int jb = 0; jb <= ib; ++jb) {
    __syncthreads();  // previous iteration done reading Vts/Ss
    if (wave == 0) {
      // V^T tile for this iteration (issued first)...
      tdm_load_2d_bf16(vs_base, Vt + (size_t)jb * 64,
                       64, DHEAD, NYR, 64, DHEAD);
      if (jb < ib) {
        // ...then next K tile into the other buffer; in-order completion
        // means tensorcnt<=1 proves K(jb) and V(jb) have landed while
        // K(jb+1) keeps streaming behind compute.
        tdm_load_2d_bf16(((jb + 1) & 1) ? ks_base1 : ks_base0,
                         K + (size_t)(jb + 1) * 64 * DHEAD,
                         DHEAD, NYR, DHEAD, DHEAD, 64);
        wait_tensor1();
      } else {
        wait_tensor0();
      }
    }
    wait_async_all();  // Q copy (first iteration); no-op afterwards
    __syncthreads();

    const __bf16* Kcur = Ks[jb & 1];

    // ---- phase 1: S(64x64) = Qs @ Kcur^T, one 16x16 tile per wave ----
    const int mt1 = wave >> 2;
    const int nt1 = wave & 3;
    v8f s = {};
    for (int kb = 0; kb < DHEAD; kb += 32) {
      v16bf aq = load_frag_a   (Qs, DHEAD, mt1 * 16, kb);
      v16bf bk = load_frag_b_nk(Kcur, DHEAD, nt1 * 16, kb);
      s = __builtin_amdgcn_wmma_f32_16x16x32_bf16(false, aq, false, bk,
                                                  (short)0, s, false, false);
    }
    // scale + strict causal mask (zero where j >= i), write bf16 S tile
    {
      const int half = lane >> 4;
      const int nloc = nt1 * 16 + (lane & 15);
      const int jg   = jb * 64 + nloc;
#pragma unroll
      for (int v = 0; v < 8; ++v) {
        const int mloc = mt1 * 16 + v + half * 8;
        const int ig   = row0 + mloc;
        float val = s[v] * scale;
        if (jg >= ig) val = 0.0f;
        Ss[mloc * 64 + nloc] = (__bf16)val;
      }
    }
    __syncthreads();

    // ---- phase 2: acc(16x128 slice of 64x512) += Ss @ V (from V^T tiles) ----
#pragma unroll
    for (int kc = 0; kc < 64; kc += 32) {
      v16bf as = load_frag_a(Ss, 64, mt2 * 16, kc);
#pragma unroll
      for (int t = 0; t < 8; ++t) {
        v16bf bv = load_frag_b_nk(Vts, 64, ncol0 + t * 16, kc);
        acc[t] = __builtin_amdgcn_wmma_f32_16x16x32_bf16(false, as, false, bv,
                                                         (short)0, acc[t],
                                                         false, false);
      }
    }
  }

  // ---- write out (f32) ----
  {
    const int half = lane >> 4;
    const int nl   = lane & 15;
#pragma unroll
    for (int t = 0; t < 8; ++t) {
      const int col = ncol0 + t * 16 + nl;
#pragma unroll
      for (int v = 0; v < 8; ++v) {
        const int row = row0 + mt2 * 16 + v + half * 8;
        out[(size_t)row * DHEAD + col] = acc[t][v];
      }
    }
  }
}

// ---------------------------------------------------------------------------
extern "C" void kernel_launch(void* const* d_in, const int* in_sizes, int n_in,
                              void* d_out, int out_size, void* d_ws, size_t ws_size,
                              hipStream_t stream) {
  const float* x  = (const float*)d_in[0];
  const float* y  = (const float*)d_in[1];
  const float* Wq = (const float*)d_in[2];
  const float* Wk = (const float*)d_in[3];
  const float* Wv = (const float*)d_in[4];
  float* out = (float*)d_out;

  __bf16* qws  = (__bf16*)d_ws;                       // [NX][D]
  __bf16* kws  = qws + (size_t)NXR * DHEAD;           // [NY][D]
  __bf16* vtws = kws + (size_t)NYR * DHEAD;           // [D][NY]  (V^T)

  dim3 gproj(NXR / 64, DHEAD / 64);
  proj_gemm_bf16<0><<<gproj, 256, 0, stream>>>(x, Wq, qws);
  proj_gemm_bf16<0><<<gproj, 256, 0, stream>>>(y, Wk, kws);
  proj_gemm_bf16<1><<<gproj, 256, 0, stream>>>(y, Wv, vtws);

  attn_stream<<<NXR / 64, 512, 0, stream>>>(qws, kws, vtws, out);
}
